// MLPPhi_44573170597935
// MI455X (gfx1250) — compile-verified, matrix operand order checked by reading, not run on hardware
//
#include <hip/hip_runtime.h>
#include <hip/hip_bf16.h>

typedef __attribute__((ext_vector_type(16))) __bf16 v16bf;
typedef __attribute__((ext_vector_type(8)))  float  v8f;

union V16 {
  v16bf v;
  __bf16 h[16];
  uint4 q[2];
};

union V8U {
  uint4 q;
  __bf16 h[8];
};

// Native f32 -> bf16 (RNE on gfx1250; should lower to v_cvt_(pk_)bf16_f32)
__device__ __forceinline__ __bf16 f2bf(float x) { return (__bf16)x; }

// Load one WMMA 16x16x32 bf16 operand tile (packed layout: tile[lane*16 + e])
__device__ __forceinline__ v16bf load_tile(const unsigned short* tile, int lane) {
  V16 t;
  const uint4* p = reinterpret_cast<const uint4*>(tile + (lane << 4));
  t.q[0] = p[0];
  t.q[1] = p[1];
  return t.v;
}

// Pack a [K, N] f32 weight into bf16 WMMA-B tiles of shape 32(K) x 16(N).
// If bias != nullptr, the spare K==K row (first pad slot) carries the bias,
// paired with a constant-1.0 element in the A matrix -> bias folded into GEMM.
// Element mapping (16-bit operands, wave32):
//   lane group g = lane/16, e<8  -> ko = e + 8g   (VGPRs 0-3)
//   e>=8                          -> ko = 16 + (e-8) + 8g (VGPRs 4-7)
__device__ void pack_weight(const float* __restrict__ src, const float* __restrict__ bias,
                            unsigned short* __restrict__ dst,
                            int K, int N, int KC, int NT, int tid, int bs) {
  int total = KC * NT * 512;
  for (int s = tid; s < total; s += bs) {
    int t   = s >> 9;
    int idx = s & 511;
    int kc  = t / NT, nt = t - kc * NT;
    int lane = idx >> 4, e = idx & 15;
    int gg = lane >> 4, np = lane & 15;
    int ko = (e < 8) ? (e + 8 * gg) : (16 + (e - 8) + 8 * gg);
    int k = kc * 32 + ko, n = nt * 16 + np;
    float v = (k < K) ? src[(size_t)k * N + n]
                      : ((bias != nullptr && k == K) ? bias[n] : 0.0f);
    dst[s] = __builtin_bit_cast(unsigned short, f2bf(v));
  }
}

// LayerNorm + ReLU over a 16x128 f32 staging buffer, all 32 lanes active:
// lanes m and m+16 each own 64 columns of row m; stats merged via shfl_xor(16).
// Output written bf16 in WMMA-A packed layout using 8x ds_store_b128 per lane.
__device__ __forceinline__ void ln_relu_pack(const float* __restrict__ Hrow,
                                             unsigned short* __restrict__ Hb,
                                             const float* __restrict__ gamma,
                                             const float* __restrict__ beta,
                                             int lane) {
  const int m = lane & 15;      // row
  const int g = lane >> 4;      // column half: [64g, 64g+64)
  const float* row = Hrow + (m << 7) + (g << 6);

  float su = 0.f, sq = 0.f;
  #pragma unroll
  for (int q = 0; q < 16; ++q) {
    float4 x = reinterpret_cast<const float4*>(row)[q];
    su += x.x + x.y + x.z + x.w;
    sq += x.x * x.x + x.y * x.y + x.z * x.z + x.w * x.w;
  }
  su += __shfl_xor(su, 16, 32);
  sq += __shfl_xor(sq, 16, 32);
  float mu  = su * 0.0078125f;
  float var = sq * 0.0078125f - mu * mu;
  float rs  = __frsqrt_rn(var + 1e-5f);

  // 8 octets of 8 consecutive columns; each octet maps to one contiguous
  // 16-byte packed slot (same target lane, e-base 0 or 8).
  #pragma unroll
  for (int oct = 0; oct < 8; ++oct) {
    const int cbase = (g << 6) + (oct << 3);
    V8U pk;
    #pragma unroll
    for (int u = 0; u < 8; ++u) {
      int c = cbase + u;
      float y = (row[(oct << 3) + u] - mu) * rs * gamma[c] + beta[c];
      y = fmaxf(y, 0.f);
      pk.h[u] = f2bf(y);
    }
    int kc = cbase >> 5, ko = cbase & 31;
    int gg    = (ko >> 3) & 1;          // octet parity -> lane group
    int ebase = (ko >> 4) << 3;         // 0 or 8
    *reinterpret_cast<uint4*>(Hb + (kc << 9) + ((m + (gg << 4)) << 4) + ebase) = pk.q;
  }
}

#define WMMA_BF16(A, B, C) \
  __builtin_amdgcn_wmma_f32_16x16x32_bf16(false, (A), false, (B), (short)0, (C), false, false)

__global__ __launch_bounds__(128) void mlp_phi_kernel(
    const float* __restrict__ W,
    const float* __restrict__ w0g, const float* __restrict__ b0g,
    const float* __restrict__ g0g, const float* __restrict__ be0g,
    const float* __restrict__ w1g, const float* __restrict__ b1g,
    const float* __restrict__ g1g, const float* __restrict__ be1g,
    const float* __restrict__ w2g, const float* __restrict__ b2g,
    const unsigned char* __restrict__ mask,
    float* __restrict__ out)
{
  __shared__ __align__(16) unsigned short w0p[8 * 512];    //  8 KB: 16(->32)x128, K=16 row = b0
  __shared__ __align__(16) unsigned short w1p[32 * 512];   // 32 KB: 128x128
  __shared__ __align__(16) unsigned short w2p[16 * 512];   // 16 KB: 128x64
  __shared__ float pg[4 * 128];                            // g0,be0,g1,be1
  __shared__ float b1s[128];
  __shared__ float b2s[64];
  __shared__ __align__(16) float Hf[4][16 * 128];          // per-wave f32 staging
  __shared__ __align__(16) unsigned short Hbp[4][4 * 512]; // per-wave A-packed bf16
  __shared__ float accW[4][64];
  __shared__ int nv;

  const int tid = threadIdx.x;
  const int bs  = blockDim.x;
  const int i   = blockIdx.x;

  if (tid == 0) nv = 0;
  __syncthreads();

  // Stage params + bf16-pack weights into WMMA tile layout (once per block).
  for (int t = tid; t < 128; t += bs) {
    pg[t]       = g0g[t];  pg[128 + t] = be0g[t];
    pg[256 + t] = g1g[t];  pg[384 + t] = be1g[t];
    b1s[t]      = b1g[t];
  }
  for (int t = tid; t < 64; t += bs) b2s[t] = b2g[t];
  pack_weight(w0g, b0g,     w0p,  16, 128, 1, 8, tid, bs);  // b0 folded into K=16 row
  pack_weight(w1g, nullptr, w1p, 128, 128, 4, 8, tid, bs);
  pack_weight(w2g, nullptr, w2p, 128,  64, 4, 4, tid, bs);

  // Valid-column count for this graph node (mask row i).
  {
    int cnt = 0;
    const unsigned char* mrow = mask + (size_t)i * 512;
    for (int t = tid; t < 512; t += bs) cnt += (mrow[t] != 0);
    atomicAdd(&nv, cnt);
  }
  __syncthreads();

  const int n_i    = nv;
  const int nTiles = (n_i + 15) >> 4;

  const int lane = tid & 31;
  const int wid  = tid >> 5;
  const int g    = lane >> 4;   // lane group (K-half selector)
  const int lm   = lane & 15;   // row (A) / column (B,C,D) within tile

  float racc[4] = {0.f, 0.f, 0.f, 0.f};
  float*          Hrow = Hf[wid];
  unsigned short* Hb   = Hbp[wid];

  for (int t = wid; t < nTiles; t += 4) {
    const int j0 = t << 4;

    // ---- Layer 0: A tile straight from global (16 rows x 16 feat, K-padded to 32).
    // K=16 element is 1.0 for lane-group 0 (pairs with b0 row in w0p) -> bias free.
    const float* wr = W + (((size_t)i * 512 + (size_t)(j0 + lm)) << 4) + (g << 3);
    float4 f0 = *reinterpret_cast<const float4*>(wr);
    float4 f1 = *reinterpret_cast<const float4*>(wr + 4);
    __builtin_prefetch(wr + 64 * 16, 0, 1);  // next tile for this wave (j0+64)
    V16 A0;
    A0.h[0] = f2bf(f0.x); A0.h[1] = f2bf(f0.y); A0.h[2] = f2bf(f0.z); A0.h[3] = f2bf(f0.w);
    A0.h[4] = f2bf(f1.x); A0.h[5] = f2bf(f1.y); A0.h[6] = f2bf(f1.z); A0.h[7] = f2bf(f1.w);
    A0.h[8] = (g == 0) ? (__bf16)1.0f : (__bf16)0.0f;  // K=16 -> 1.0 pairs with b0 row
    #pragma unroll
    for (int e = 9; e < 16; ++e) A0.h[e] = (__bf16)0.0f;

    #pragma unroll
    for (int nt = 0; nt < 8; ++nt) {
      v16bf B = load_tile(w0p + (nt << 9), lane);
      v8f c = {};                                     // inline-0 C operand
      c = WMMA_BF16(A0.v, B, c);
      #pragma unroll
      for (int r = 0; r < 8; ++r)
        Hrow[((r + (g << 3)) << 7) + (nt << 4) + lm] = c[r];
    }
    asm volatile("s_wait_dscnt 0" ::: "memory");
    ln_relu_pack(Hrow, Hb, pg + 0, pg + 128, lane);
    asm volatile("s_wait_dscnt 0" ::: "memory");

    // ---- Layer 1: 128 -> 128 (+ b1, needed before LN)
    V16 A1[4];
    #pragma unroll
    for (int kc = 0; kc < 4; ++kc) A1[kc].v = load_tile(Hb + (kc << 9), lane);

    #pragma unroll
    for (int nt = 0; nt < 8; ++nt) {
      float bb = b1s[(nt << 4) + lm];
      v8f c;
      #pragma unroll
      for (int r = 0; r < 8; ++r) c[r] = bb;          // + b1
      #pragma unroll
      for (int kc = 0; kc < 4; ++kc) {
        v16bf B = load_tile(w1p + (((kc << 3) + nt) << 9), lane);
        c = WMMA_BF16(A1[kc].v, B, c);
      }
      #pragma unroll
      for (int r = 0; r < 8; ++r)
        Hrow[((r + (g << 3)) << 7) + (nt << 4) + lm] = c[r];
    }
    asm volatile("s_wait_dscnt 0" ::: "memory");
    ln_relu_pack(Hrow, Hb, pg + 256, pg + 384, lane);
    asm volatile("s_wait_dscnt 0" ::: "memory");

    // ---- Layer 2: 128 -> 64, C = 0 (b2 hoisted to writeout), masked row-sum
    V16 A2[4];
    #pragma unroll
    for (int kc = 0; kc < 4; ++kc) A2[kc].v = load_tile(Hb + (kc << 9), lane);

    #pragma unroll
    for (int nt = 0; nt < 4; ++nt) {
      v8f c = {};                                     // inline-0 C operand
      #pragma unroll
      for (int kc = 0; kc < 4; ++kc) {
        v16bf B = load_tile(w2p + (((kc << 2) + nt) << 9), lane);
        c = WMMA_BF16(A2[kc].v, B, c);
      }
      float s = 0.f;
      #pragma unroll
      for (int r = 0; r < 8; ++r) {
        int m = r + (g << 3);
        if (j0 + m < n_i) s += c[r];                  // mask invalid rows
      }
      racc[nt] += s;
    }
  }

  // Deterministic reduction: fold lane halves, one slot per wave, fixed-order sum.
  #pragma unroll
  for (int nt = 0; nt < 4; ++nt) {
    float v = racc[nt] + __shfl_xor(racc[nt], 16, 32);
    if (lane < 16) accW[wid][(nt << 4) + lane] = v;
  }
  __syncthreads();
  if (tid < 64)
    out[(size_t)i * 64 + tid] = accW[0][tid] + accW[1][tid] + accW[2][tid] + accW[3][tid]
                              + (float)n_i * b2s[tid];   // Sum_valid(b2) = n_i * b2
}

extern "C" void kernel_launch(void* const* d_in, const int* in_sizes, int n_in,
                              void* d_out, int out_size, void* d_ws, size_t ws_size,
                              hipStream_t stream) {
  const float* W   = (const float*)d_in[0];
  const float* w0  = (const float*)d_in[1];
  const float* b0  = (const float*)d_in[2];
  const float* g0  = (const float*)d_in[3];
  const float* be0 = (const float*)d_in[4];
  const float* w1  = (const float*)d_in[5];
  const float* b1  = (const float*)d_in[6];
  const float* g1  = (const float*)d_in[7];
  const float* be1 = (const float*)d_in[8];
  const float* w2  = (const float*)d_in[9];
  const float* b2  = (const float*)d_in[10];
  const unsigned char* mask = (const unsigned char*)d_in[11];
  float* out = (float*)d_out;

  const int n_sum = in_sizes[11] / 512;  // mask is [N_sum, 512]
  hipLaunchKernelGGL(mlp_phi_kernel, dim3(n_sum), dim3(128), 0, stream,
                     W, w0, b0, g0, be0, w1, b1, g1, be1, w2, b2, mask, out);
}
